// CrossAttention_35673998361109
// MI455X (gfx1250) — compile-verified, matrix operand order checked by reading, not run on hardware
//
#include <hip/hip_runtime.h>
#include <cstdint>
#include <cstddef>

typedef __bf16 v16bf __attribute__((ext_vector_type(16)));
typedef float  v8f   __attribute__((ext_vector_type(8)));

#define DEV static __device__ __forceinline__
#define NEG_HUGE (-3.0e38f)

DEV unsigned short f2bf(float f) {
  unsigned int u = __float_as_uint(f);
  u += 0x7FFFu + ((u >> 16) & 1u);          // round-to-nearest-even
  return (unsigned short)(u >> 16);
}

DEV v8f wmma_bf16(v16bf a, v16bf b, v8f c) {
  return __builtin_amdgcn_wmma_f32_16x16x32_bf16(false, a, false, b, (short)0, c,
                                                 false, false);
}

// gfx1250 async global->LDS copy, 16B per lane, tracked by ASYNCcnt.
DEV void async_b128(unsigned lds_off, const void* gptr) {
  unsigned long long g = (unsigned long long)(size_t)gptr;
  asm volatile("global_load_async_to_lds_b128 %0, %1, off"
               :: "v"(lds_off), "v"(g) : "memory");
}
DEV void wait_async0() { asm volatile("s_wait_asynccnt 0x0" ::: "memory"); }

// LDS byte offset of a generic (shared) pointer: low 32 bits of the flat addr.
DEV unsigned lds_off(const void* p) { return (unsigned)(size_t)p; }

// XOR-lane shuffle via ds_swizzle (group-of-32: and=31, or=0, xor=XM).
template <int XM>
DEV float swz_xor(float v) {
  return __int_as_float(
      __builtin_amdgcn_ds_swizzle(__float_as_int(v), (XM << 10) | 31));
}
DEV float redmax16(float v) {           // max across 16-lane half
  v = fmaxf(v, swz_xor<8>(v));
  v = fmaxf(v, swz_xor<4>(v));
  v = fmaxf(v, swz_xor<2>(v));
  v = fmaxf(v, swz_xor<1>(v));
  return v;
}
DEV float redsum16(float v) {           // sum across 16-lane half
  v += swz_xor<8>(v);
  v += swz_xor<4>(v);
  v += swz_xor<2>(v);
  v += swz_xor<1>(v);
  return v;
}

// Load one 16x32 bf16 operand fragment from an LDS tile (row stride in shorts).
// ISA A-layout: lane&15 = matrix row; lane>>4 selects K+8; VGPRs 0-3 hold
// K[kb..kb+7], VGPRs 4-7 hold K[kb+16..kb+23].  B uses the same layout on the
// transposed (N-major, K-contiguous) operand.
DEV v16bf ldfrag(const unsigned short* base, int stride, int lane) {
  const unsigned short* p = base + (lane & 15) * stride + ((lane >> 4) << 3);
  v16bf f;
  ((uint4*)&f)[0] = *(const uint4*)(p);
  ((uint4*)&f)[1] = *(const uint4*)(p + 16);
  return f;
}

// ---------------------------------------------------------------- LayerNorm
__global__ __launch_bounds__(256)
void ln_bf16_kernel(const float* __restrict__ x, const float* __restrict__ g,
                    const float* __restrict__ b, unsigned short* __restrict__ out) {
  __shared__ float s1[256], s2[256];
  const int row = blockIdx.x, tid = threadIdx.x;
  const float4 v = ((const float4*)(x + (size_t)row * 1024))[tid];
  float sum = v.x + v.y + v.z + v.w;
  float sq  = v.x * v.x + v.y * v.y + v.z * v.z + v.w * v.w;
  s1[tid] = sum; s2[tid] = sq;
  __syncthreads();
  for (int s = 128; s > 0; s >>= 1) {
    if (tid < s) { s1[tid] += s1[tid + s]; s2[tid] += s2[tid + s]; }
    __syncthreads();
  }
  const float mu  = s1[0] * (1.0f / 1024.0f);
  const float var = s2[0] * (1.0f / 1024.0f) - mu * mu;
  const float rs  = rsqrtf(var + 1e-5f);
  const float4 gg = ((const float4*)g)[tid];
  const float4 bb = ((const float4*)b)[tid];
  ushort4 h;
  h.x = f2bf((v.x - mu) * rs * gg.x + bb.x);
  h.y = f2bf((v.y - mu) * rs * gg.y + bb.y);
  h.z = f2bf((v.z - mu) * rs * gg.z + bb.z);
  h.w = f2bf((v.w - mu) * rs * gg.w + bb.w);
  ((ushort4*)(out + (size_t)row * 1024))[tid] = h;
}

// ----------------------- fp32 W[K][N] -> bf16 Wt[N][K]  (N = power of two)
__global__ __launch_bounds__(256)
void convT_bf16_kernel(const float* __restrict__ W, unsigned short* __restrict__ Wt,
                       int K, int lgN) {
  size_t idx = (size_t)blockIdx.x * 256 + threadIdx.x;
  int k = (int)(idx >> lgN);
  int n = (int)(idx & (((size_t)1 << lgN) - 1));
  if (k < K) Wt[(size_t)n * K + k] = f2bf(W[idx]);
}

// ------------------------------------------------------------- WMMA GEMM
// C[M][N] = alpha * A[M][K] @ Bt[N][K]^T ;  A,Bt bf16 row-major over K.
// Block tile 128x128, 8 waves, each wave 32x64.  Double-buffered async LDS.
// MODE: 0 = bf16 store, 1 = f32 store, 2 = f32 accumulate,
//       3 = KV split: cols 0-63 -> bf16 K [m][64]; cols 64-127 -> bf16
//           V transposed [b][d][j] (b = m>>11, j = m&2047).
#define LPAD 40
#define TBUF (128 * LPAD)           // shorts per tile buffer
template <int MODE, int K>
__global__ __launch_bounds__(256)
void gemm_kernel(const unsigned short* __restrict__ A,
                 const unsigned short* __restrict__ Bt, void* __restrict__ Cv,
                 unsigned short* __restrict__ Vt, int N, float alpha) {
  __shared__ __attribute__((aligned(16))) unsigned short As[2 * TBUF];
  __shared__ __attribute__((aligned(16))) unsigned short Bs[2 * TBUF];
  const int tid = threadIdx.x, lane = tid & 31, wave = tid >> 5;
  const int m0 = blockIdx.y * 128, n0 = blockIdx.x * 128;
  const int wm = (wave & 3) * 32, wn = (wave >> 2) * 64;
  const int r0 = tid >> 2, c8 = (tid & 3) * 8;
  const unsigned short* gA0 = A  + (size_t)(m0 + r0) * K + c8;
  const unsigned short* gA1 = gA0 + (size_t)64 * K;
  const unsigned short* gB0 = Bt + (size_t)(n0 + r0) * K + c8;
  const unsigned short* gB1 = gB0 + (size_t)64 * K;
  const unsigned lA = lds_off(As + r0 * LPAD + c8);
  const unsigned lB = lds_off(Bs + r0 * LPAD + c8);
  auto stage = [&](int k0, int buf) {
    const unsigned bo = (unsigned)buf * (TBUF * 2);           // bytes
    async_b128(lA + bo,                 gA0 + k0);
    async_b128(lA + bo + 64 * LPAD * 2, gA1 + k0);
    async_b128(lB + bo,                 gB0 + k0);
    async_b128(lB + bo + 64 * LPAD * 2, gB1 + k0);
  };
  v8f acc[2][4] = {};
  stage(0, 0);
  int buf = 0;
  for (int k0 = 0; k0 < K; k0 += 32, buf ^= 1) {
    wait_async0();
    __syncthreads();
    if (k0 + 32 < K) stage(k0 + 32, buf ^ 1);
    const unsigned short* Ab = As + buf * TBUF;
    const unsigned short* Bb = Bs + buf * TBUF;
    v16bf a0 = ldfrag(Ab + (wm + 0)  * LPAD, LPAD, lane);
    v16bf a1 = ldfrag(Ab + (wm + 16) * LPAD, LPAD, lane);
#pragma unroll
    for (int j = 0; j < 4; ++j) {
      v16bf bj = ldfrag(Bb + (wn + 16 * j) * LPAD, LPAD, lane);
      acc[0][j] = wmma_bf16(a0, bj, acc[0][j]);
      acc[1][j] = wmma_bf16(a1, bj, acc[1][j]);
    }
  }
  const int col = lane & 15, half = lane >> 4;
#pragma unroll
  for (int i = 0; i < 2; ++i)
#pragma unroll
    for (int j = 0; j < 4; ++j)
#pragma unroll
      for (int r = 0; r < 8; ++r) {
        int m = m0 + wm + 16 * i + r + 8 * half;
        int n = n0 + wn + 16 * j + col;
        float v = acc[i][j][r] * alpha;
        size_t idx = (size_t)m * N + n;
        if (MODE == 1) {
          ((float*)Cv)[idx] = v;
        } else if (MODE == 2) {
          ((float*)Cv)[idx] += v;
        } else if (MODE == 3) {
          int bb = m >> 11, jj = m & 2047;
          if (n < 64)
            ((unsigned short*)Cv)[(size_t)m * 64 + n] = f2bf(v);
          else
            Vt[((size_t)(bb * 64 + (n - 64))) * 2048 + jj] = f2bf(v);
        } else {
          ((unsigned short*)Cv)[idx] = f2bf(v);
        }
      }
}

// ------------------------------------- FF1: fused val/gate GEMM + SwiGLU
// H[M][4096] = silu(A@W1t[4096+n]) * (A@W1t[n]) ; tile 128x64, K template.
#define BBUF (64 * LPAD)
template <int K>
__global__ __launch_bounds__(256)
void ff1_swiglu_kernel(const unsigned short* __restrict__ A,
                       const unsigned short* __restrict__ W1t,
                       unsigned short* __restrict__ H) {
  __shared__ __attribute__((aligned(16))) unsigned short As[2 * TBUF];
  __shared__ __attribute__((aligned(16))) unsigned short Bv[2 * BBUF];
  __shared__ __attribute__((aligned(16))) unsigned short Bg[2 * BBUF];
  const int tid = threadIdx.x, lane = tid & 31, wave = tid >> 5;
  const int m0 = blockIdx.y * 128, n0 = blockIdx.x * 64;
  const int wm = (wave & 3) * 32, wn = (wave >> 2) * 32;
  const int r0 = tid >> 2, c8 = (tid & 3) * 8;
  const unsigned short* gA0 = A + (size_t)(m0 + r0) * K + c8;
  const unsigned short* gA1 = gA0 + (size_t)64 * K;
  const unsigned short* gV  = W1t + (size_t)(n0 + r0) * K + c8;
  const unsigned short* gG  = W1t + (size_t)(4096 + n0 + r0) * K + c8;
  const unsigned lA = lds_off(As + r0 * LPAD + c8);
  const unsigned lV = lds_off(Bv + r0 * LPAD + c8);
  const unsigned lG = lds_off(Bg + r0 * LPAD + c8);
  auto stage = [&](int k0, int buf) {
    async_b128(lA + buf * (TBUF * 2),                 gA0 + k0);
    async_b128(lA + buf * (TBUF * 2) + 64 * LPAD * 2, gA1 + k0);
    async_b128(lV + buf * (BBUF * 2),                 gV + k0);
    async_b128(lG + buf * (BBUF * 2),                 gG + k0);
  };
  v8f av[2][2] = {}, ag[2][2] = {};
  stage(0, 0);
  int buf = 0;
  for (int k0 = 0; k0 < K; k0 += 32, buf ^= 1) {
    wait_async0();
    __syncthreads();
    if (k0 + 32 < K) stage(k0 + 32, buf ^ 1);
    const unsigned short* Ab = As + buf * TBUF;
    const unsigned short* Vb = Bv + buf * BBUF;
    const unsigned short* Gb = Bg + buf * BBUF;
    v16bf a0 = ldfrag(Ab + (wm + 0)  * LPAD, LPAD, lane);
    v16bf a1 = ldfrag(Ab + (wm + 16) * LPAD, LPAD, lane);
    v16bf v0 = ldfrag(Vb + (wn + 0)  * LPAD, LPAD, lane);
    v16bf v1 = ldfrag(Vb + (wn + 16) * LPAD, LPAD, lane);
    v16bf g0 = ldfrag(Gb + (wn + 0)  * LPAD, LPAD, lane);
    v16bf g1 = ldfrag(Gb + (wn + 16) * LPAD, LPAD, lane);
    av[0][0] = wmma_bf16(a0, v0, av[0][0]);
    av[0][1] = wmma_bf16(a0, v1, av[0][1]);
    av[1][0] = wmma_bf16(a1, v0, av[1][0]);
    av[1][1] = wmma_bf16(a1, v1, av[1][1]);
    ag[0][0] = wmma_bf16(a0, g0, ag[0][0]);
    ag[0][1] = wmma_bf16(a0, g1, ag[0][1]);
    ag[1][0] = wmma_bf16(a1, g0, ag[1][0]);
    ag[1][1] = wmma_bf16(a1, g1, ag[1][1]);
  }
  const int col = lane & 15, half = lane >> 4;
#pragma unroll
  for (int i = 0; i < 2; ++i)
#pragma unroll
    for (int j = 0; j < 2; ++j)
#pragma unroll
      for (int r = 0; r < 8; ++r) {
        int m = m0 + wm + 16 * i + r + 8 * half;
        int n = n0 + wn + 16 * j + col;
        float val = av[i][j][r], gate = ag[i][j][r];
        float res = val * gate / (1.0f + __expf(-gate));   // silu(gate)*val
        H[(size_t)m * 4096 + n] = f2bf(res);
      }
}

// --------------------------------------------------- Flash attention (MQA)
// qbuf: [b*2048][1024] bf16 (pre-scaled), kbuf: [b*2048][64] bf16,
// vtbuf: [b][64][2048] bf16 (V pre-transposed by the KV GEMM),
// mask: fp32 additive [b][2048][2048], attnout: [b*2048][1024] bf16.
#define APAD 72
#define KVBUF (64 * APAD)
__global__ __launch_bounds__(128)
void attn_kernel(const unsigned short* __restrict__ qbuf,
                 const unsigned short* __restrict__ kbuf,
                 const unsigned short* __restrict__ vtbuf,
                 const float* __restrict__ mask,
                 unsigned short* __restrict__ attnout) {
  __shared__ __attribute__((aligned(16))) unsigned short qs[KVBUF];
  __shared__ __attribute__((aligned(16))) unsigned short ksm[2 * KVBUF];
  __shared__ __attribute__((aligned(16))) unsigned short vts[2 * KVBUF];
  __shared__ __attribute__((aligned(16))) unsigned short ps[4 * 16 * APAD];
  const int tid = threadIdx.x, lane = tid & 31, wave = tid >> 5;
  const int b = blockIdx.y >> 4, h = blockIdx.y & 15;
  const int i0 = blockIdx.x * 64;
  const size_t rb = (size_t)b * 2048;
  const int sr = tid >> 3, sc = (tid & 7) * 8;
  // stage Q tile asynchronously (64 rows x 64 cols of this head)
  {
    const unsigned lq = lds_off(qs + sr * APAD + sc);
    const unsigned short* gq = qbuf + (rb + i0 + sr) * 1024 + h * 64 + sc;
#pragma unroll
    for (int s = 0; s < 4; ++s)
      async_b128(lq + s * 16 * APAD * 2, gq + (size_t)s * 16 * 1024);
  }
  const unsigned lk = lds_off(ksm + sr * APAD + sc);
  const unsigned lv = lds_off(vts + sr * APAD + sc);
  const unsigned short* gk = kbuf + (rb + sr) * 64 + sc;
  const unsigned short* gv = vtbuf + ((size_t)(b * 64 + sr)) * 2048 + sc;
  auto stageKV = [&](int j0, int buf) {
    const unsigned bo = (unsigned)buf * (KVBUF * 2);
#pragma unroll
    for (int s = 0; s < 4; ++s) {
      async_b128(lk + bo + s * 16 * APAD * 2, gk + (size_t)(j0 + 16 * s) * 64);
      async_b128(lv + bo + s * 16 * APAD * 2, gv + (size_t)(16 * s) * 2048 + j0);
    }
  };
  v8f o[4] = {};
  float mrow[8], lrow[8];
#pragma unroll
  for (int r = 0; r < 8; ++r) { mrow[r] = NEG_HUGE; lrow[r] = 0.0f; }
  const int col = lane & 15, half = lane >> 4;

  stageKV(0, 0);
  int buf = 0;
  for (int j0 = 0; j0 < 2048; j0 += 64, buf ^= 1) {
    wait_async0();
    __syncthreads();
    if (j0 + 64 < 2048) stageKV(j0 + 64, buf ^ 1);
    const unsigned short* Kb = ksm + buf * KVBUF;
    const unsigned short* Vb = vts + buf * KVBUF;
    // S = Q @ K^T  (16 x 64 per wave)
    v8f s[4] = {};
#pragma unroll
    for (int kk = 0; kk < 64; kk += 32) {
      v16bf aq = ldfrag(qs + wave * 16 * APAD + kk, APAD, lane);
#pragma unroll
      for (int jt = 0; jt < 4; ++jt) {
        v16bf bk = ldfrag(Kb + jt * 16 * APAD + kk, APAD, lane);
        s[jt] = wmma_bf16(aq, bk, s[jt]);
      }
    }
    // additive mask + online softmax
    const float* mp = mask + (rb + i0 + wave * 16 + 8 * half) * 2048 + j0 + col;
    float mcur[8];
#pragma unroll
    for (int r = 0; r < 8; ++r) mcur[r] = NEG_HUGE;
#pragma unroll
    for (int jt = 0; jt < 4; ++jt)
#pragma unroll
      for (int r = 0; r < 8; ++r) {
        float sv = s[jt][r] + mp[(size_t)r * 2048 + jt * 16];
        s[jt][r] = sv;
        mcur[r] = fmaxf(mcur[r], sv);
      }
#pragma unroll
    for (int r = 0; r < 8; ++r) {
      mcur[r] = redmax16(mcur[r]);
      float mn = fmaxf(mrow[r], mcur[r]);
      float sc2 = __expf(mrow[r] - mn);
      mrow[r] = mn;
      lrow[r] *= sc2;
#pragma unroll
      for (int dt = 0; dt < 4; ++dt) o[dt][r] *= sc2;
    }
    float psum[8] = {0, 0, 0, 0, 0, 0, 0, 0};
    unsigned short* pw = ps + wave * 16 * APAD;
#pragma unroll
    for (int jt = 0; jt < 4; ++jt)
#pragma unroll
      for (int r = 0; r < 8; ++r) {
        float p = __expf(s[jt][r] - mrow[r]);
        psum[r] += p;
        pw[(r + 8 * half) * APAD + jt * 16 + col] = f2bf(p);
      }
#pragma unroll
    for (int r = 0; r < 8; ++r) lrow[r] += redsum16(psum[r]);
    __syncthreads();
    // O += P @ V
#pragma unroll
    for (int kk = 0; kk < 64; kk += 32) {
      v16bf ap = ldfrag(pw + kk, APAD, lane);
#pragma unroll
      for (int dt = 0; dt < 4; ++dt) {
        v16bf bv = ldfrag(Vb + dt * 16 * APAD + kk, APAD, lane);
        o[dt] = wmma_bf16(ap, bv, o[dt]);
      }
    }
  }
#pragma unroll
  for (int r = 0; r < 8; ++r) {
    float inv = 1.0f / lrow[r];
    size_t orow = rb + i0 + wave * 16 + r + 8 * half;
#pragma unroll
    for (int dt = 0; dt < 4; ++dt)
      attnout[orow * 1024 + h * 64 + dt * 16 + col] = f2bf(o[dt][r] * inv);
  }
}

// ------------------------------------------------------------------- host
extern "C" void kernel_launch(void* const* d_in, const int* in_sizes, int n_in,
                              void* d_out, int out_size, void* d_ws, size_t ws_size,
                              hipStream_t stream) {
  (void)in_sizes; (void)n_in; (void)out_size; (void)ws_size;
  const float* x     = (const float*)d_in[0];
  const float* ctx   = (const float*)d_in[1];
  const float* mask  = (const float*)d_in[2];
  const float* ln_g  = (const float*)d_in[3];
  const float* ln_b  = (const float*)d_in[4];
  const float* cln_g = (const float*)d_in[5];
  const float* cln_b = (const float*)d_in[6];
  const float* Wq    = (const float*)d_in[7];
  const float* Wkv   = (const float*)d_in[8];
  const float* Wout  = (const float*)d_in[9];
  const float* W1    = (const float*)d_in[10];
  const float* W2    = (const float*)d_in[11];

  char* ws = (char*)d_ws;
  size_t off = 0;
  auto alloc = [&](size_t bytes) {
    size_t o = off; off += (bytes + 255) & ~(size_t)255; return o;
  };
  const size_t M = 4096;  // b * n (== b * j)
  unsigned short* xn      = (unsigned short*)(ws + alloc(M * 1024 * 2));
  unsigned short* cn      = (unsigned short*)(ws + alloc(M * 1024 * 2));
  unsigned short* Wq_t    = (unsigned short*)(ws + alloc(1024 * 1024 * 2));
  unsigned short* Wkv_t   = (unsigned short*)(ws + alloc(128 * 1024 * 2));
  unsigned short* Wout_t  = (unsigned short*)(ws + alloc(1024 * 1024 * 2));
  unsigned short* W1_t    = (unsigned short*)(ws + alloc((size_t)8192 * 1024 * 2));
  unsigned short* W2_t    = (unsigned short*)(ws + alloc((size_t)1024 * 4096 * 2));
  unsigned short* qbuf    = (unsigned short*)(ws + alloc(M * 1024 * 2));
  unsigned short* kbuf    = (unsigned short*)(ws + alloc(M * 64 * 2));
  unsigned short* vtbuf   = (unsigned short*)(ws + alloc(M * 64 * 2));
  unsigned short* attnout = (unsigned short*)(ws + alloc(M * 1024 * 2));
  unsigned short* hbuf    = (unsigned short*)(ws + alloc(M * 4096 * 2));

  // 1) layernorms -> bf16
  ln_bf16_kernel<<<4096, 256, 0, stream>>>(x, ln_g, ln_b, xn);
  ln_bf16_kernel<<<4096, 256, 0, stream>>>(ctx, cln_g, cln_b, cn);
  // 2) weights -> transposed bf16 (lgN = log2 of column count)
  convT_bf16_kernel<<<(1024 * 1024) / 256, 256, 0, stream>>>(Wq, Wq_t, 1024, 10);
  convT_bf16_kernel<<<(1024 * 128) / 256, 256, 0, stream>>>(Wkv, Wkv_t, 1024, 7);
  convT_bf16_kernel<<<(1024 * 1024) / 256, 256, 0, stream>>>(Wout, Wout_t, 1024, 10);
  convT_bf16_kernel<<<(1024 * 8192) / 256, 256, 0, stream>>>(W1, W1_t, 1024, 13);
  convT_bf16_kernel<<<(4096 * 1024) / 256, 256, 0, stream>>>(W2, W2_t, 4096, 10);
  // 3) projections (scale 1/sqrt(64) folded into q); KV writes K + V^T
  gemm_kernel<0, 1024><<<dim3(8, 32), 256, 0, stream>>>(xn, Wq_t, qbuf, nullptr, 1024, 0.125f);
  gemm_kernel<3, 1024><<<dim3(1, 32), 256, 0, stream>>>(cn, Wkv_t, kbuf, vtbuf, 128, 1.0f);
  // 4) flash attention
  attn_kernel<<<dim3(32, 32), 128, 0, stream>>>(qbuf, kbuf, vtbuf, mask, attnout);
  // 5) output projection -> fp32 d_out
  gemm_kernel<1, 1024><<<dim3(8, 32), 256, 0, stream>>>(attnout, Wout_t, d_out, nullptr, 1024, 1.0f);
  // 6) feedforward: fused SwiGLU, then accumulate into d_out
  ff1_swiglu_kernel<1024><<<dim3(64, 32), 256, 0, stream>>>(xn, W1_t, hbuf);
  gemm_kernel<2, 4096><<<dim3(8, 32), 256, 0, stream>>>(hbuf, W2_t, d_out, nullptr, 1024, 1.0f);
}
